// ImageRNN_72816875536480
// MI455X (gfx1250) — matrix operand order repaired and checked by reading, not compiled, not run on hardware
//
#include <hip/hip_runtime.h>
#include <stdint.h>

typedef __attribute__((ext_vector_type(16))) _Float16 v16h;
typedef __attribute__((ext_vector_type(8)))  float    v8f;

#define B_TOTAL 16384
#define T_STEPS 28
#define I_DIM   128
#define H_DIM   256
#define O_DIM   10
#define KT_N    12          // K = I+H = 384 -> 12 k-tiles of 32
#define ROWS    64          // batch rows per workgroup
#define LSTR    392         // weight row stride (halves): 196 dwords % 64 = 4 -> conflict-free
#define HSTR    264         // h row stride (halves):      132 dwords % 64 = 4 -> conflict-free
#define XSTR    132         // x row stride (floats):      132 dwords % 64 = 4 -> conflict-free

// Async copy 16B global -> LDS (gfx1250, tracked by ASYNCcnt)
static __device__ __forceinline__ void async_copy16(uint32_t lds_off, const float* gaddr) {
    asm volatile("global_load_async_to_lds_b128 %0, %1, off"
                 :: "v"(lds_off), "v"(gaddr)
                 : "memory");
}
static __device__ __forceinline__ void wait_async0() {
    asm volatile("s_wait_asynccnt 0x0" ::: "memory");
}

__global__ __launch_bounds__(256, 1)
void rnn_fused(const float* __restrict__ X,
               const float* __restrict__ W_ih,
               const float* __restrict__ b_ih,
               const float* __restrict__ W_hh,
               const float* __restrict__ b_hh,
               const float* __restrict__ W_fc,
               const float* __restrict__ b_fc,
               float* __restrict__ out)
{
    // G[n][k] combined weights (f16): k<128 -> W_ih[n][k], k>=128 -> W_hh[n][k-128]
    __shared__ __align__(16) _Float16 w_lds[H_DIM * LSTR];      // 200,704 B
    __shared__ __align__(16) _Float16 h_lds[ROWS * HSTR];       //  33,792 B
    __shared__ __align__(16) float    xbuf [2][ROWS * XSTR];    //  67,584 B (async double buffer)

    const int tid   = threadIdx.x;
    const int wave  = tid >> 5;
    const int lane  = tid & 31;
    const int g     = lane >> 4;      // lane half-group (WMMA layout)
    const int ln    = lane & 15;
    const int strip = wave & 3;       // 16-row strip within the 64-row tile
    const int nhalf = wave >> 2;      // 0 -> n-tiles 0..7, 1 -> n-tiles 8..15
    const int row0  = blockIdx.x * ROWS;

    // ---- kick off async staging of x_0 immediately (overlaps weight conversion) ----
    #pragma unroll
    for (int i = 0; i < 8; ++i) {                  // 2048 x 16B chunks / 256 threads
        int idx = tid + 256 * i;
        int r = idx >> 5, c4 = idx & 31;
        const float* gp = X + ((size_t)(row0 + r) * T_STEPS + 0) * I_DIM + c4 * 4;
        async_copy16((uint32_t)(uintptr_t)&xbuf[0][r * XSTR + c4 * 4], gp);
    }

    // ---- load combined weight matrix into LDS as f16 (once) ----
    {
        const float4* wi4 = (const float4*)W_ih;   // 256x128 f32 = 8192 float4
        #pragma unroll 4
        for (int i = 0; i < 32; ++i) {
            int idx = tid + 256 * i;
            int r = idx >> 5, c4 = idx & 31;
            float4 v = wi4[idx];
            _Float16* d = &w_lds[r * LSTR + c4 * 4];
            d[0] = (_Float16)v.x; d[1] = (_Float16)v.y;
            d[2] = (_Float16)v.z; d[3] = (_Float16)v.w;
        }
        const float4* wh4 = (const float4*)W_hh;   // 256x256 f32 = 16384 float4
        #pragma unroll 4
        for (int i = 0; i < 64; ++i) {
            int idx = tid + 256 * i;
            int r = idx >> 6, c4 = idx & 63;
            float4 v = wh4[idx];
            _Float16* d = &w_lds[r * LSTR + 128 + c4 * 4];
            d[0] = (_Float16)v.x; d[1] = (_Float16)v.y;
            d[2] = (_Float16)v.z; d[3] = (_Float16)v.w;
        }
    }
    // ---- zero h (h0 = 0) ----
    #pragma unroll
    for (int i = 0; i < 16; ++i) {                 // 64 rows * 64 chunks of 4 halves
        int idx = tid + 256 * i;
        int r = idx >> 6, c4 = idx & 63;
        _Float16* d = &h_lds[r * HSTR + c4 * 4];
        d[0] = (_Float16)0.f; d[1] = (_Float16)0.f;
        d[2] = (_Float16)0.f; d[3] = (_Float16)0.f;
    }

    // per-lane bias for each of this wave's 8 n-tiles
    float biasv[8];
    #pragma unroll
    for (int nt = 0; nt < 8; ++nt) {
        int nc = nhalf * 128 + nt * 16 + ln;
        biasv[nt] = b_ih[nc] + b_hh[nc];
    }

    const int arow = strip * 16 + ln;              // this lane's A-matrix row

    for (int t = 0; t < T_STEPS; ++t) {
        wait_async0();           // our async chunks of x_t have landed in LDS
        __syncthreads();         // everyone's async done + previous h writes visible

        // ---- immediately prefetch x_{t+1} into the other buffer: overlaps the
        //      entire step (A-build + barriers + WMMA compute). Its previous
        //      readers finished in step t-1, ordered by the barrier above. ----
        if (t + 1 < T_STEPS) {
            float* dst = xbuf[(t + 1) & 1];
            #pragma unroll
            for (int i = 0; i < 8; ++i) {
                int idx = tid + 256 * i;
                int r = idx >> 5, c4 = idx & 31;
                const float* gp = X + ((size_t)(row0 + r) * T_STEPS + (t + 1)) * I_DIM + c4 * 4;
                async_copy16((uint32_t)(uintptr_t)&dst[r * XSTR + c4 * 4], gp);
            }
        }

        // ---- preload all 12 A fragments into registers ----
        // 16-bit A 16x32 layout: lane group g: halves 0..7 -> K=8g+j, halves 8..15 -> K=16+8g+j
        v16h afrag[KT_N];
        {
            const float* xp = &xbuf[t & 1][arow * XSTR];   // x part (f32 -> f16 convert)
            #pragma unroll
            for (int kt = 0; kt < 4; ++kt) {
                const float* p = xp + kt * 32 + g * 8;
                v16h a;
                #pragma unroll
                for (int j = 0; j < 8; ++j) {
                    a[j]     = (_Float16)p[j];
                    a[8 + j] = (_Float16)p[16 + j];
                }
                afrag[kt] = a;
            }
            const _Float16* hp = &h_lds[arow * HSTR];      // h part (already f16)
            #pragma unroll
            for (int kt = 0; kt < 8; ++kt) {
                const uint4* p = (const uint4*)(hp + kt * 32 + g * 8);
                union { v16h v; uint4 q[2]; } u;
                u.q[0] = p[0];
                u.q[1] = p[2];
                afrag[4 + kt] = u.v;
            }
        }
        __syncthreads();         // all x/h reads done -> h writable

        // ---- 8 n-tiles processed as 4 independent pairs (2 WMMA chains for ILP) ----
        #pragma unroll 1
        for (int ntp = 0; ntp < 4; ++ntp) {
            int nt0 = 2 * ntp;
            int nc0 = nhalf * 128 + nt0 * 16 + ln;         // lane's output column, tile 0
            int nc1 = nc0 + 16;                            // tile 1
            float b0 = biasv[nt0], b1 = biasv[nt0 + 1];
            v8f acc0 = { b0, b0, b0, b0, b0, b0, b0, b0 };
            v8f acc1 = { b1, b1, b1, b1, b1, b1, b1, b1 };
            // 16-bit B 32x16 layout: lane group g, col = ln: halves 0..7 -> K=8g+j, halves 8..15 -> K=16+8g+j
            const _Float16* wp0 = &w_lds[nc0 * LSTR + g * 8];
            const _Float16* wp1 = wp0 + 16 * LSTR;
            #pragma unroll
            for (int kt = 0; kt < KT_N; ++kt) {
                union { v16h v; uint4 q[2]; } ub0, ub1;
                const uint4* p0 = (const uint4*)(wp0 + kt * 32);
                const uint4* p1 = (const uint4*)(wp1 + kt * 32);
                ub0.q[0] = p0[0]; ub0.q[1] = p0[2];
                ub1.q[0] = p1[0]; ub1.q[1] = p1[2];
                acc0 = __builtin_amdgcn_wmma_f32_16x16x32_f16(
                           false, afrag[kt], false, ub0.v, (short)0, acc0, false, false);
                acc1 = __builtin_amdgcn_wmma_f32_16x16x32_f16(
                           false, afrag[kt], false, ub1.v, (short)0, acc1, false, false);
            }
            // C layout: vgpr v -> row m = strip*16 + 8g + v
            int mrow = strip * 16 + 8 * g;
            #pragma unroll
            for (int v = 0; v < 8; ++v) {
                float x0 = acc0[v]; x0 = x0 > 0.f ? x0 : 0.f;
                float x1 = acc1[v]; x1 = x1 > 0.f ? x1 : 0.f;
                h_lds[(mrow + v) * HSTR + nc0] = (_Float16)x0;
                h_lds[(mrow + v) * HSTR + nc1] = (_Float16)x1;
            }
        }
    }
    __syncthreads();

    // ---- FC head: out[b][o] = h[b][:] . W_fc[o][:] + b_fc[o] ----
    {
        int r  = tid & 63;
        int og = tid >> 6;                       // 0..3
        int o0 = og, o1 = og + 4, o2 = og + 8;   // o2 valid for og < 2
        float s0 = 0.f, s1 = 0.f, s2 = 0.f;
        const _Float16* hrow = &h_lds[r * HSTR];
        #pragma unroll 4
        for (int k = 0; k < H_DIM; ++k) {
            float hv = (float)hrow[k];
            s0 += hv * W_fc[o0 * H_DIM + k];
            s1 += hv * W_fc[o1 * H_DIM + k];
            if (og < 2) s2 += hv * W_fc[o2 * H_DIM + k];
        }
        int orow = (row0 + r) * O_DIM;
        out[orow + o0] = s0 + b_fc[o0];
        out[orow + o1] = s1 + b_fc[o1];
        if (og < 2) out[orow + o2] = s2 + b_fc[o2];
    }
}

extern "C" void kernel_launch(void* const* d_in, const int* in_sizes, int n_in,
                              void* d_out, int out_size, void* d_ws, size_t ws_size,
                              hipStream_t stream) {
    const float* X    = (const float*)d_in[0];
    const float* W_ih = (const float*)d_in[1];
    const float* b_ih = (const float*)d_in[2];
    const float* W_hh = (const float*)d_in[3];
    const float* b_hh = (const float*)d_in[4];
    const float* W_fc = (const float*)d_in[5];
    const float* b_fc = (const float*)d_in[6];
    float* out = (float*)d_out;

    dim3 grid(B_TOTAL / ROWS);   // 256 workgroups, one 64-row batch strip each
    dim3 block(256);             // 8 wave32s
    hipLaunchKernelGGL(rnn_fused, grid, block, 0, stream,
                       X, W_ih, b_ih, W_hh, b_hh, W_fc, b_fc, out);
}
